// DeltaResidualBlock_11063835755016
// MI455X (gfx1250) — matrix operand order repaired and checked by reading, not compiled, not run on hardware
//
#include <hip/hip_runtime.h>
#include <math.h>

#define DIM   2048
#define TOK   16384          // 4 * 4096 tokens
#define EPSV  1e-8f

typedef __attribute__((ext_vector_type(4)))  float  v4f;
typedef __attribute__((ext_vector_type(8)))  float  v8f;
typedef __attribute__((ext_vector_type(8)))  __bf16 v8bf;
typedef __attribute__((ext_vector_type(16))) __bf16 v16bf;

// ---------------------------------------------------------------------------
// K1: k_raw[m,e] = sum_d H[m,d] * Wk[e,d]   via V_WMMA_F32_16X16X32_BF16
// Block tile 128x128, 8 waves as 2(M) x 4(N); wave tile 64x32 -> 4x2 frags.
// fp32 tiles converted to bf16 on the fly; fp32 accumulation in C.
// ---------------------------------------------------------------------------
#define BM 128
#define BN 128
#define BK 32
#define PITCH 40   // bf16 elems per LDS row: 80 B -> conflict-free, 16B-aligned

static __device__ __forceinline__ v8bf cvt8(v4f a, v4f b) {
  v8bf r;
  r[0] = (__bf16)a.x; r[1] = (__bf16)a.y; r[2] = (__bf16)a.z; r[3] = (__bf16)a.w;
  r[4] = (__bf16)b.x; r[5] = (__bf16)b.y; r[6] = (__bf16)b.z; r[7] = (__bf16)b.w;
  return r;
}

static __device__ __forceinline__ v16bf cat16(v8bf lo, v8bf hi) {
  return __builtin_shufflevector(lo, hi, 0, 1, 2, 3, 4, 5, 6, 7,
                                         8, 9, 10, 11, 12, 13, 14, 15);
}

__global__ __launch_bounds__(256) void gemm_kraw_wmma(
    const float* __restrict__ H, const float* __restrict__ Wk,
    float* __restrict__ C) {
  __shared__ __bf16 ldsA[BM][PITCH];
  __shared__ __bf16 ldsB[BN][PITCH];

  const int tid    = threadIdx.x;
  const int lane   = tid & 31;
  const int wave   = tid >> 5;
  const int laneLo = lane & 15;
  const int laneHi = lane >> 4;       // 0 or 1
  const int wm     = wave >> 2;       // 0..1  (M)
  const int wn     = wave & 3;        // 0..3  (N)

  const int m0 = blockIdx.y * BM;
  const int n0 = blockIdx.x * BN;

  v8f acc[4][2];
#pragma unroll
  for (int i = 0; i < 4; ++i)
#pragma unroll
    for (int j = 0; j < 2; ++j)
#pragma unroll
      for (int r = 0; r < 8; ++r) acc[i][j][r] = 0.0f;

  // Global tile loads: 2 threads per row, 16 contiguous fp32 (64 B) each.
  const int lrow  = tid >> 1;          // 0..127
  const int lhalf = (tid & 1) * 16;    // k-element offset: 0 or 16

  const float* pA = &H [(size_t)(m0 + lrow) * DIM + lhalf];
  const float* pB = &Wk[(size_t)(n0 + lrow) * DIM + lhalf];

  v4f ra[4], rb[4];
#pragma unroll
  for (int q = 0; q < 4; ++q) {        // preload stage 0
    ra[q] = *(const v4f*)(pA + 4 * q);
    rb[q] = *(const v4f*)(pB + 4 * q);
  }

  for (int kc = 0; kc < DIM; kc += BK) {
    // Stage current regs into LDS as bf16
    *(v8bf*)&ldsA[lrow][lhalf]     = cvt8(ra[0], ra[1]);
    *(v8bf*)&ldsA[lrow][lhalf + 8] = cvt8(ra[2], ra[3]);
    *(v8bf*)&ldsB[lrow][lhalf]     = cvt8(rb[0], rb[1]);
    *(v8bf*)&ldsB[lrow][lhalf + 8] = cvt8(rb[2], rb[3]);
    __syncthreads();

    // Software pipeline: issue next stage's global loads before compute
    if (kc + BK < DIM) {
#pragma unroll
      for (int q = 0; q < 4; ++q) {
        ra[q] = *(const v4f*)(pA + kc + BK + 4 * q);
        rb[q] = *(const v4f*)(pB + kc + BK + 4 * q);
      }
    }

    // Fragments in contiguous k order (same for A and B):
    //  A (16x32): lane m holds contiguous K[16*laneHi .. +15]
    //  B (32x16): lane n holds contiguous K[16*laneHi .. +15]
    v16bf afr[4], bfr[2];
#pragma unroll
    for (int mi = 0; mi < 4; ++mi) {
      const __bf16* a0 = &ldsA[wm * 64 + mi * 16 + laneLo][16 * laneHi];
      afr[mi] = cat16(*(const v8bf*)a0, *(const v8bf*)(a0 + 8));
    }
#pragma unroll
    for (int ni = 0; ni < 2; ++ni) {
      const __bf16* b0 = &ldsB[wn * 32 + ni * 16 + laneLo][16 * laneHi];
      bfr[ni] = cat16(*(const v8bf*)b0, *(const v8bf*)(b0 + 8));
    }
#pragma unroll
    for (int mi = 0; mi < 4; ++mi)
#pragma unroll
      for (int ni = 0; ni < 2; ++ni)
        acc[mi][ni] = __builtin_amdgcn_wmma_f32_16x16x32_bf16(
            /*neg_a=*/false, afr[mi], /*neg_b=*/false, bfr[ni],
            /*c_mod=*/(short)0, acc[mi][ni],
            /*reuse_a=*/false, /*reuse_b=*/false);
    __syncthreads();
  }

  // C/D layout: VGPR r -> (M = r, lanes 0-15) and (M = r+8, lanes 16-31)
#pragma unroll
  for (int mi = 0; mi < 4; ++mi) {
#pragma unroll
    for (int ni = 0; ni < 2; ++ni) {
      const int col = n0 + wn * 32 + ni * 16 + laneLo;
#pragma unroll
      for (int r = 0; r < 8; ++r) {
        const int row = m0 + wm * 64 + mi * 16 + r + 8 * laneHi;
        C[(size_t)row * DIM + col] = acc[mi][ni][r];
      }
    }
  }
}

// ---------------------------------------------------------------------------
// K2: per-token epilogue (in-place over k_raw stored in d_out)
// h_next = h + beta * (k_raw/(|k_raw|+eps)) * (v - (k_raw.h)/(|k_raw|+eps))
// ---------------------------------------------------------------------------
__global__ __launch_bounds__(256) void epilogue_kernel(
    const float* __restrict__ H,
    const float* __restrict__ Wb, const float* __restrict__ bb,
    const float* __restrict__ Wv, const float* __restrict__ bv,
    float* __restrict__ C, float* __restrict__ betas) {
  __shared__ v4f  red[256];
  __shared__ float bcast;

  const int t   = blockIdx.x;
  const int tid = threadIdx.x;
  const float* hr = H + (size_t)t * DIM;
  float*       kr = C + (size_t)t * DIM;

  float kv[8], hv8[8];
  float skk = 0.f, skh = 0.f, shv = 0.f, shb = 0.f;
#pragma unroll
  for (int j = 0; j < 8; ++j) {
    const int idx = j * 256 + tid;
    const float kx = kr[idx];
    const float hx = hr[idx];
    kv[j] = kx; hv8[j] = hx;
    skk = fmaf(kx, kx, skk);
    skh = fmaf(kx, hx, skh);
    shv = fmaf(hx, Wv[idx], shv);
    shb = fmaf(hx, Wb[idx], shb);
  }
  red[tid] = (v4f){skk, skh, shv, shb};
  __syncthreads();
#pragma unroll
  for (int s = 128; s > 0; s >>= 1) {   // fixed-order tree -> deterministic
    if (tid < s) red[tid] += red[tid + s];
    __syncthreads();
  }
  if (tid == 0) {
    const v4f r    = red[0];
    const float inv  = 1.0f / (sqrtf(r.x) + EPSV);
    const float vval = r.z + bv[0];
    const float beta = 2.0f / (1.0f + expf(-(r.w + bb[0])));
    const float kd   = r.y * inv;
    bcast    = beta * inv * (vval - kd);
    betas[t] = beta;
  }
  __syncthreads();
  const float cf = bcast;
#pragma unroll
  for (int j = 0; j < 8; ++j) {
    const int idx = j * 256 + tid;
    kr[idx] = fmaf(cf, kv[j], hv8[j]);
  }
}

// ---------------------------------------------------------------------------
// K3: beta_mean (deterministic single-block reduction)
// ---------------------------------------------------------------------------
__global__ __launch_bounds__(256) void beta_mean_kernel(
    const float* __restrict__ betas, float* __restrict__ outv) {
  __shared__ float s[256];
  float a = 0.f;
  for (int i = threadIdx.x; i < TOK; i += 256) a += betas[i];
  s[threadIdx.x] = a;
  __syncthreads();
#pragma unroll
  for (int st = 128; st > 0; st >>= 1) {
    if (threadIdx.x < st) s[threadIdx.x] += s[threadIdx.x + st];
    __syncthreads();
  }
  if (threadIdx.x == 0) outv[0] = s[0] / (float)TOK;
}

// ---------------------------------------------------------------------------
extern "C" void kernel_launch(void* const* d_in, const int* in_sizes, int n_in,
                              void* d_out, int out_size, void* d_ws, size_t ws_size,
                              hipStream_t stream) {
  const float* H  = (const float*)d_in[0];  // (4,4096,2048)
  const float* Wk = (const float*)d_in[1];  // (2048,2048)
  const float* Wb = (const float*)d_in[2];  // (2048,)
  const float* bb = (const float*)d_in[3];  // (1,)
  const float* Wv = (const float*)d_in[4];  // (2048,)
  const float* bv = (const float*)d_in[5];  // (1,)
  float* out   = (float*)d_out;             // [TOK*DIM] h_next, then [1] beta_mean
  float* betas = (float*)d_ws;              // TOK floats = 64 KB scratch

  dim3 g1(DIM / BN, TOK / BM);              // 16 x 128 blocks
  gemm_kraw_wmma<<<g1, 256, 0, stream>>>(H, Wk, out);
  epilogue_kernel<<<TOK, 256, 0, stream>>>(H, Wb, bb, Wv, bv, out, betas);
  beta_mean_kernel<<<1, 256, 0, stream>>>(betas, out + (size_t)TOK * DIM);
}
